// RetNet_23613730193588
// MI455X (gfx1250) — matrix operand order repaired and checked by reading, not compiled, not run on hardware
//
#include <hip/hip_runtime.h>
#include <cstdint>

typedef __bf16 bf16;
typedef __attribute__((ext_vector_type(16))) __bf16 v16bf;
typedef __attribute__((ext_vector_type(8)))  __bf16 v8bf;
typedef __attribute__((ext_vector_type(8)))  float  v8f;

#define BB 2
#define SS 2048
#define DD 256
#define HH 4
#define DKk 64
#define VV 512
#define DVv 128
#define FF 1024
#define LL 3
#define MM (BB*SS)   // 4096 tokens

static __device__ __forceinline__ v8f wmma_bf16(v16bf a, v16bf b, v8f c) {
    return __builtin_amdgcn_wmma_f32_16x16x32_bf16(false, a, false, b, (short)0, c, false, false);
}

// A-fragment (16x32 bf16): lane holds row m=lane%16; k = 16*half .. 16*half+15 contiguous
static __device__ __forceinline__ v16bf load_a_frag(const bf16* __restrict__ row, int koff0) {
    v8bf lo = *(const v8bf*)(row + koff0);
    v8bf hi = *(const v8bf*)(row + koff0 + 8);
    v16bf a;
#pragma unroll
    for (int i = 0; i < 8; ++i) { a[i] = lo[i]; a[8 + i] = hi[i]; }
    return a;
}

// ---------------- weight packing: fp32 (K x N) -> bf16 tile-swizzled B fragments ----------------
// tile = (n/16)*(K/32) + (k/32); lane = (n%16) + 16*((k%32)/16); elem = k%16
__global__ void pack_w_kernel(const float* __restrict__ W, bf16* __restrict__ P, int K, int N) {
    int idx = blockIdx.x * blockDim.x + threadIdx.x;
    if (idx >= K * N) return;
    int k = idx / N, n = idx % N;
    int tile = (n >> 4) * (K >> 5) + (k >> 5);
    int lane = (n & 15) + 16 * ((k & 31) >> 4);
    int e = k & 15;
    P[(size_t)tile * 512 + lane * 16 + e] = (bf16)W[idx];
}

// ---------------- LayerNorm: one wave per row of 256, out bf16 ----------------
__global__ void ln_kernel(const float* __restrict__ x, const float* __restrict__ w,
                          const float* __restrict__ b, bf16* __restrict__ out) {
    int lane = threadIdx.x & 31;
    int row = blockIdx.x * 8 + (threadIdx.x >> 5);
    const float* xr = x + (size_t)row * DD;
    float v[8];
    float s = 0.f;
#pragma unroll
    for (int i = 0; i < 8; ++i) { v[i] = xr[lane + 32 * i]; s += v[i]; }
#pragma unroll
    for (int m = 1; m < 32; m <<= 1) s += __shfl_xor(s, m, 32);
    float mu = s * (1.f / DD);
    float var = 0.f;
#pragma unroll
    for (int i = 0; i < 8; ++i) { float d = v[i] - mu; var += d * d; }
#pragma unroll
    for (int m = 1; m < 32; m <<= 1) var += __shfl_xor(var, m, 32);
    float inv = rsqrtf(var * (1.f / DD) + 1e-5f);
    bf16* orow = out + (size_t)row * DD;
#pragma unroll
    for (int i = 0; i < 8; ++i) {
        int c = lane + 32 * i;
        orow[c] = (bf16)((v[i] - mu) * inv * w[c] + b[c]);
    }
}

// ---------------- Generic WMMA GEMM: C(M x N) = A(M x K, bf16 rowmajor) @ Bpacked ----------------
// Each wave: 16 rows x 64 cols (4 WMMA accumulators; A-frag reused 4x).
// Epilogues: 0=store f32, 1=+resid f32, 2=+bias,gelu->bf16, 3=+bias+resid f32
template <int EPI>
__global__ __launch_bounds__(128) void gemm_kernel(const bf16* __restrict__ A, const bf16* __restrict__ Bp,
                                                   int K, int N,
                                                   float* __restrict__ outF, bf16* __restrict__ outB,
                                                   const float* __restrict__ bias,
                                                   const float* __restrict__ resid) {
    const int lane = threadIdx.x & 31;
    const int wave = threadIdx.x >> 5;
    const int m0 = (blockIdx.x * 4 + wave) * 16;
    const int half = lane >> 4;
    const int col = lane & 15;
    const int kt = K >> 5;
    v8f c[4];
#pragma unroll
    for (int j = 0; j < 4; ++j) c[j] = (v8f){};
    const bf16* arow = A + (size_t)(m0 + col) * K;
    const bf16* bp[4];
#pragma unroll
    for (int j = 0; j < 4; ++j)
        bp[j] = Bp + (size_t)(blockIdx.y * 4 + j) * kt * 512 + lane * 16;
    for (int kc = 0; kc < kt; ++kc) {
        v16bf a = load_a_frag(arow + kc * 32, 16 * half);
        if (kc + 2 < kt) {  // gfx1250 global_prefetch path for the B panel
            __builtin_prefetch(bp[0] + (kc + 2) * 512, 0, 3);
            __builtin_prefetch(bp[2] + (kc + 2) * 512, 0, 3);
        }
#pragma unroll
        for (int j = 0; j < 4; ++j) {
            v16bf b = *(const v16bf*)(bp[j] + kc * 512);
            c[j] = wmma_bf16(a, b, c[j]);
        }
    }
#pragma unroll
    for (int j = 0; j < 4; ++j) {
        int n = (blockIdx.y * 4 + j) * 16 + col;
#pragma unroll
        for (int r = 0; r < 8; ++r) {
            int row = m0 + r + 8 * half;
            float v = c[j][r];
            size_t i = (size_t)row * N + n;
            if (EPI == 0) {
                outF[i] = v;
            } else if (EPI == 1) {
                outF[i] = v + resid[i];
            } else if (EPI == 2) {
                v += bias[n];
                v = 0.5f * v * (1.f + erff(v * 0.70710678f));
                outB[i] = (bf16)v;
            } else {
                outF[i] = v + bias[n] + resid[i];
            }
        }
    }
}

// ---------------- theta shift (rotary) on q,k; write bf16 (b,h,s,dk); k scaled by DK^-0.5 ----------------
__global__ void theta_kernel(const float* __restrict__ q, const float* __restrict__ k,
                             bf16* __restrict__ qr, bf16* __restrict__ kr) {
    int lane = threadIdx.x & 31;                    // pair index j
    int idx = blockIdx.x * 8 + (threadIdx.x >> 5);  // (b,s,h)
    int h = idx & (HH - 1);
    int bs = idx >> 2;             // b*S + s
    int s = bs & (SS - 1);
    int b = bs >> 11;
    float ang = (float)s * __powf(10000.f, -(float)lane / 31.f);
    float sn, cs;
    __sincosf(ang, &sn, &cs);
    size_t src = (size_t)bs * DD + h * DKk + 2 * lane;
    size_t dst = ((size_t)(b * HH + h) * SS + s) * DKk + 2 * lane;
    float q1 = q[src], q2 = q[src + 1];
    qr[dst]     = (bf16)(q1 * cs - q2 * sn);
    qr[dst + 1] = (bf16)(q2 * cs + q1 * sn);
    float k1 = k[src] * 0.125f, k2 = k[src + 1] * 0.125f;  // DK^-0.5
    kr[dst]     = (bf16)(k1 * cs - k2 * sn);
    kr[dst + 1] = (bf16)(k2 * cs + k1 * sn);
}

// ---------------- v transpose: (b,s,h*DV+dv) f32 -> (b,h,dv,s) bf16 ----------------
__global__ void vtrans_kernel(const float* __restrict__ v, bf16* __restrict__ vT) {
    int idx = blockIdx.x * blockDim.x + threadIdx.x;
    if (idx >= BB * HH * DVv * SS) return;
    int s = idx & (SS - 1);
    int dv = (idx >> 11) & (DVv - 1);
    int h = (idx >> 18) & (HH - 1);
    int b = idx >> 20;
    vT[idx] = (bf16)v[((size_t)b * SS + s) * VV + h * DVv + dv];
}

// ---------------- retention: one wave per (b,h, 16-row s-block) ----------------
__global__ __launch_bounds__(32) void attn_kernel(const bf16* __restrict__ qr, const bf16* __restrict__ kr,
                                                  const bf16* __restrict__ vT, float* __restrict__ o) {
    __shared__ bf16 sqk[16 * 32];
    const int lane = threadIdx.x;
    const int half = lane >> 4;
    const int col = lane & 15;
    const int sb = blockIdx.x & 127;
    const int bh = blockIdx.x >> 7;
    const int h = bh & (HH - 1);
    const int b = bh >> 2;
    const int s0 = sb * 16;
    const float d = logf(1.f - exp2f(-5.f - (float)h));  // per-head decay (negative)
    // analytic inverse row norm: rowsum(m) = (1 - e^{(m+1)d}) / (1 - e^d)
    float invn[8];
    const float one_m_ed = 1.f - expf(d);
#pragma unroll
    for (int r = 0; r < 8; ++r) {
        int m = s0 + r + 8 * half;
        invn[r] = rsqrtf((1.f - expf((float)(m + 1) * d)) / one_m_ed);
    }
    // q fragments for full DK=64
    const bf16* arow = qr + ((size_t)bh * SS + s0 + col) * DKk;
    v16bf aq0 = load_a_frag(arow, 16 * half);
    v16bf aq1 = load_a_frag(arow + 32, 16 * half);

    v8f ov[8]; float den[8];
#pragma unroll
    for (int i = 0; i < 8; ++i) { ov[i] = (v8f){}; den[i] = 0.f; }

    const int nch = (s0 + 15) / 32 + 1;  // causal chunk count (t in steps of 32)
    for (int ch = 0; ch < nch; ++ch) {
        const int tb = ch * 32;
        // qk tiles (16x16 each) over K = DK = 64
        v8f qk0 = {}, qk1 = {};
        const bf16* kb = kr + ((size_t)bh * SS + tb + col) * DKk + 16 * half;
        v16bf b00 = *(const v16bf*)(kb);
        v16bf b01 = *(const v16bf*)(kb + 32);
        v16bf b10 = *(const v16bf*)(kb + 16 * DKk);
        v16bf b11 = *(const v16bf*)(kb + 16 * DKk + 32);
        qk0 = wmma_bf16(aq0, b00, qk0); qk0 = wmma_bf16(aq1, b01, qk0);
        qk1 = wmma_bf16(aq0, b10, qk1); qk1 = wmma_bf16(aq1, b11, qk1);
        // mask + denom accumulation + stage masked qk (bf16) to LDS for re-layout
#pragma unroll
        for (int tt = 0; tt < 2; ++tt) {
            int t_abs = tb + tt * 16 + col;
#pragma unroll
            for (int r = 0; r < 8; ++r) {
                int m_abs = s0 + r + 8 * half;
                int diff = m_abs - t_abs;
                float val = tt ? qk1[r] : qk0[r];
                float msk = (diff >= 0) ? expf((float)diff * d) * invn[r] : 0.f;
                float qm = val * msk;
                den[r] += fabsf(qm);
                sqk[(r + 8 * half) * 32 + tt * 16 + col] = (bf16)qm;
            }
        }
        __syncthreads();
        const bf16* sp = &sqk[col * 32 + 16 * half];
        v8bf lo = *(const v8bf*)sp;
        v8bf hi = *(const v8bf*)(sp + 8);
        v16bf aqk;
#pragma unroll
        for (int i = 0; i < 8; ++i) { aqk[i] = lo[i]; aqk[8 + i] = hi[i]; }
        __syncthreads();
        // o += qk(16x32) @ v(32x128): 8 WMMAs across DV tiles
        const bf16* vb = vT + ((size_t)bh * DVv + col) * SS + tb + 16 * half;
#pragma unroll
        for (int dvt = 0; dvt < 8; ++dvt) {
            v16bf bv = *(const v16bf*)(vb + (size_t)dvt * 16 * SS);
            ov[dvt] = wmma_bf16(aqk, bv, ov[dvt]);
        }
    }
    // denom: reduce across 16 columns within half, clip at 1, divide
#pragma unroll
    for (int r = 0; r < 8; ++r) {
#pragma unroll
        for (int m = 1; m < 16; m <<= 1) den[r] += __shfl_xor(den[r], m, 32);
        den[r] = 1.f / fmaxf(den[r], 1.f);
    }
    // rms norm over DV=128 then store (b, s, h*DV+dv)
    float ssq[8];
#pragma unroll
    for (int r = 0; r < 8; ++r) ssq[r] = 0.f;
#pragma unroll
    for (int dvt = 0; dvt < 8; ++dvt)
#pragma unroll
        for (int r = 0; r < 8; ++r) { float t = ov[dvt][r] * den[r]; ov[dvt][r] = t; ssq[r] += t * t; }
#pragma unroll
    for (int r = 0; r < 8; ++r) {
#pragma unroll
        for (int m = 1; m < 16; m <<= 1) ssq[r] += __shfl_xor(ssq[r], m, 32);
        ssq[r] = rsqrtf(ssq[r] * (1.f / DVv) + 1e-6f);
    }
#pragma unroll
    for (int dvt = 0; dvt < 8; ++dvt)
#pragma unroll
        for (int r = 0; r < 8; ++r) {
            int row = s0 + r + 8 * half;
            o[((size_t)b * SS + row) * VV + h * DVv + dvt * 16 + col] = ov[dvt][r] * ssq[r];
        }
}

// ---------------- gate: og = silu(g) * o, bf16 ----------------
__global__ void gate_kernel(const float* __restrict__ g, const float* __restrict__ o, bf16* __restrict__ og) {
    int idx = blockIdx.x * blockDim.x + threadIdx.x;
    if (idx >= MM * VV) return;
    float gv = g[idx];
    og[idx] = (bf16)(gv / (1.f + expf(-gv)) * o[idx]);
}

// ==================================================================================
extern "C" void kernel_launch(void* const* d_in, const int* in_sizes, int n_in,
                              void* d_out, int out_size, void* d_ws, size_t ws_size,
                              hipStream_t stream) {
    const float* x0    = (const float*)d_in[0];
    const float* Wq    = (const float*)d_in[1];
    const float* Wk    = (const float*)d_in[2];
    const float* Wv    = (const float*)d_in[3];
    const float* Wg    = (const float*)d_in[4];
    const float* Wo    = (const float*)d_in[5];
    const float* ln1w  = (const float*)d_in[6];
    const float* ln1b  = (const float*)d_in[7];
    const float* ln2w  = (const float*)d_in[8];
    const float* ln2b  = (const float*)d_in[9];
    const float* W1    = (const float*)d_in[10];
    const float* b1    = (const float*)d_in[11];
    const float* W2    = (const float*)d_in[12];
    const float* b2    = (const float*)d_in[13];
    float* out = (float*)d_out;

    char* base = (char*)d_ws;
    size_t off = 0;
    auto alloc = [&](size_t bytes) -> char* {
        char* p = base + off;
        off += (bytes + 255) & ~(size_t)255;
        return p;
    };
    bf16* wq_p = (bf16*)alloc((size_t)LL * DD * DD * 2);
    bf16* wk_p = (bf16*)alloc((size_t)LL * DD * DD * 2);
    bf16* wv_p = (bf16*)alloc((size_t)LL * DD * VV * 2);
    bf16* wg_p = (bf16*)alloc((size_t)LL * DD * VV * 2);
    bf16* wo_p = (bf16*)alloc((size_t)LL * VV * DD * 2);
    bf16* w1_p = (bf16*)alloc((size_t)LL * DD * FF * 2);
    bf16* w2_p = (bf16*)alloc((size_t)LL * FF * DD * 2);
    bf16* lnout = (bf16*)alloc((size_t)MM * DD * 2);
    float* qbuf = (float*)alloc((size_t)MM * DD * 4);
    float* kbuf = (float*)alloc((size_t)MM * DD * 4);
    float* vbuf = (float*)alloc((size_t)MM * VV * 4);
    float* gbuf = (float*)alloc((size_t)MM * VV * 4);
    bf16* qrb  = (bf16*)alloc((size_t)MM * DD * 2);  // (b,h,s,dk)
    bf16* krb  = (bf16*)alloc((size_t)MM * DD * 2);
    bf16* vTb  = (bf16*)alloc((size_t)MM * VV * 2);  // (b,h,dv,s)
    float* obuf = (float*)alloc((size_t)MM * VV * 4);
    bf16* ogb  = (bf16*)alloc((size_t)MM * VV * 2);
    float* ybuf = (float*)alloc((size_t)MM * DD * 4);
    bf16* hbuf = (bf16*)alloc((size_t)MM * DD * 2);
    bf16* f1b  = (bf16*)alloc((size_t)MM * FF * 2);
    float* xA  = (float*)alloc((size_t)MM * DD * 4);
    float* xB  = (float*)alloc((size_t)MM * DD * 4);

    auto packW = [&](const float* W, bf16* P, int K, int N) {
        pack_w_kernel<<<(K * N + 255) / 256, 256, 0, stream>>>(W, P, K, N);
    };
    for (int i = 0; i < LL; ++i) {
        packW(Wq + (size_t)i * DD * DD, wq_p + (size_t)i * DD * DD, DD, DD);
        packW(Wk + (size_t)i * DD * DD, wk_p + (size_t)i * DD * DD, DD, DD);
        packW(Wv + (size_t)i * DD * VV, wv_p + (size_t)i * DD * VV, DD, VV);
        packW(Wg + (size_t)i * DD * VV, wg_p + (size_t)i * DD * VV, DD, VV);
        packW(Wo + (size_t)i * VV * DD, wo_p + (size_t)i * VV * DD, VV, DD);
        packW(W1 + (size_t)i * DD * FF, w1_p + (size_t)i * DD * FF, DD, FF);
        packW(W2 + (size_t)i * FF * DD, w2_p + (size_t)i * FF * DD, FF, DD);
    }

    const float* x_cur = x0;
    for (int i = 0; i < LL; ++i) {
        float* x_next = (i == 2) ? out : (i == 0 ? xA : xB);
        // ln1
        ln_kernel<<<MM / 8, 256, 0, stream>>>(x_cur, ln1w + i * DD, ln1b + i * DD, lnout);
        // projections (wave tile 16x64 => grid.y = N/64)
        dim3 blk(128);
        gemm_kernel<0><<<dim3(MM / 64, DD / 64), blk, 0, stream>>>(lnout, wq_p + (size_t)i * DD * DD, DD, DD, qbuf, nullptr, nullptr, nullptr);
        gemm_kernel<0><<<dim3(MM / 64, DD / 64), blk, 0, stream>>>(lnout, wk_p + (size_t)i * DD * DD, DD, DD, kbuf, nullptr, nullptr, nullptr);
        gemm_kernel<0><<<dim3(MM / 64, VV / 64), blk, 0, stream>>>(lnout, wv_p + (size_t)i * DD * VV, DD, VV, vbuf, nullptr, nullptr, nullptr);
        gemm_kernel<0><<<dim3(MM / 64, VV / 64), blk, 0, stream>>>(lnout, wg_p + (size_t)i * DD * VV, DD, VV, gbuf, nullptr, nullptr, nullptr);
        // rotary + layouts
        theta_kernel<<<(BB * SS * HH) / 8, 256, 0, stream>>>(qbuf, kbuf, qrb, krb);
        vtrans_kernel<<<(MM * VV + 255) / 256, 256, 0, stream>>>(vbuf, vTb);
        // retention
        attn_kernel<<<BB * HH * (SS / 16), 32, 0, stream>>>(qrb, krb, vTb, obuf);
        // gate + output projection (+ residual x)
        gate_kernel<<<(MM * VV + 255) / 256, 256, 0, stream>>>(gbuf, obuf, ogb);
        gemm_kernel<1><<<dim3(MM / 64, DD / 64), blk, 0, stream>>>(ogb, wo_p + (size_t)i * VV * DD, VV, DD, ybuf, nullptr, nullptr, x_cur);
        // FFN
        ln_kernel<<<MM / 8, 256, 0, stream>>>(ybuf, ln2w + i * DD, ln2b + i * DD, hbuf);
        gemm_kernel<2><<<dim3(MM / 64, FF / 64), blk, 0, stream>>>(hbuf, w1_p + (size_t)i * DD * FF, DD, FF, nullptr, f1b, b1 + i * FF, nullptr);
        gemm_kernel<3><<<dim3(MM / 64, DD / 64), blk, 0, stream>>>(f1b, w2_p + (size_t)i * FF * DD, FF, DD, x_next, nullptr, b2 + i * DD, ybuf);
        x_cur = x_next;
    }
    (void)in_sizes; (void)n_in; (void)out_size; (void)ws_size;
}
